// CausalSelfAttention_46608985096884
// MI455X (gfx1250) — matrix operand order repaired and checked by reading, not compiled, not run on hardware
//
#include <hip/hip_runtime.h>

// ---------------------------------------------------------------------------
// Causal self-attention fwd for MI455X (gfx1250, wave32, WMMA).
// fp32 emulated with bf16x3 split WMMA:  A*B ~= Ah*Bh + Ah*Bl + Al*Bh.
// GEMMs use GLOBAL_LOAD_ASYNC_TO_LDS_B128 double-buffered LDS staging
// (ASYNCcnt pipeline) feeding v_wmma_f32_16x16x32_bf16.
// ---------------------------------------------------------------------------

typedef __attribute__((ext_vector_type(16))) __bf16 v16bf;
typedef __attribute__((ext_vector_type(8)))  __bf16 v8bf;
typedef __attribute__((ext_vector_type(8)))  float  v8f;

constexpr int  Bc  = 2;
constexpr int  Tc  = 2048;
constexpr int  Dc  = 2048;
constexpr int  Hc  = 16;
constexpr int  HDc = 128;
constexpr long BTDc = (long)Bc * Tc * Dc;   // 8388608
constexpr long DDc  = (long)Dc * Dc;        // 4194304

// ------------------------------- device helpers ----------------------------

__device__ __forceinline__ v8f wmma_bf16(v16bf a, v16bf b, v8f c) {
  return __builtin_amdgcn_wmma_f32_16x16x32_bf16(false, a, false, b,
                                                 (short)0, c, false, false);
}

// A fragment: 16x32 bf16 tile at `base`, row stride `stride` (elements).
__device__ __forceinline__ v16bf ldA(const __bf16* base, long stride, int lane) {
  const int ln = lane & 15, hf = lane >> 4;
  const __bf16* p = base + (long)ln * stride + 16 * hf;
  v16bf r;
  ((v8bf*)&r)[0] = *(const v8bf*)(p);
  ((v8bf*)&r)[1] = *(const v8bf*)(p + 8);
  return r;
}

// B fragment: 32x16, source row n at base+n*stride holds K-contiguous data.
__device__ __forceinline__ v16bf ldB(const __bf16* base, long stride, int lane) {
  const int ln = lane & 15, hf = lane >> 4;
  return *(const v16bf*)(base + (long)ln * stride + 16 * hf);
}

// ------------------------------- split kernels -----------------------------

__global__ void __launch_bounds__(256)
split_kernel(const float* __restrict__ src, __bf16* __restrict__ hi,
             __bf16* __restrict__ lo, long n) {
  long i = (long)blockIdx.x * 256 + threadIdx.x;
  if (i >= n) return;
  float f = src[i];
  __bf16 h = (__bf16)f;
  hi[i] = h;
  lo[i] = (__bf16)(f - (float)h);
}

// v: (B,T,H,HD) f32  ->  vt hi/lo: (B,H,HD,T) bf16 (transposed for PV B-frags)
__global__ void __launch_bounds__(256)
splitT_kernel(const float* __restrict__ v, __bf16* __restrict__ th,
              __bf16* __restrict__ tl) {
  long i = (long)blockIdx.x * 256 + threadIdx.x;
  if (i >= BTDc) return;
  int  din = (int)(i % Dc);
  long bt  = i / Dc;
  int  hh  = din / HDc;
  int  dd  = din % HDc;
  int  b   = (int)(bt / Tc);
  int  t   = (int)(bt % Tc);
  float f  = v[i];
  __bf16 h = (__bf16)f;
  long o = (((long)b * Hc + hh) * HDc + dd) * Tc + t;
  th[o] = h;
  tl[o] = (__bf16)(f - (float)h);
}

// ------------------------------- RoPE --------------------------------------

__global__ void __launch_bounds__(256)
rope_kernel(float* __restrict__ x, const float* __restrict__ cs,
            const float* __restrict__ sn) {
  long i = (long)blockIdx.x * 256 + threadIdx.x;   // over B*T*H*(HD/2)
  if (i >= (long)Bc * Tc * Hc * (HDc / 2)) return;
  int  pr   = (int)(i & 63);
  long rest = i >> 6;
  int  hh   = (int)(rest % Hc);
  long bt   = rest / Hc;               // b*T + t
  int  t    = (int)(bt % Tc);
  long base = bt * Dc + (long)hh * HDc + 2 * pr;
  float c  = cs[t * 64 + pr];
  float s  = sn[t * 64 + pr];
  float xr = x[base], xi = x[base + 1];
  x[base]     = xr * c - xi * s;
  x[base + 1] = xr * s + xi * c;
}

// ------------------------------- GEMM (C = A * W^T) ------------------------
// Block = 256 thr (8 waves, 2Mx4N), block tile 128x128, k-step 32.
// 128x32 hi/lo tiles of A and W are staged to LDS with
// GLOBAL_LOAD_ASYNC_TO_LDS_B128, double-buffered on ASYNCcnt:
// 8 async ops per thread per stage; "s_wait_asynccnt 8" retires the
// previous stage while the next one is in flight.

__global__ void __launch_bounds__(256)
gemm_bf16x3(const __bf16* __restrict__ Ah, const __bf16* __restrict__ Al,
            const __bf16* __restrict__ Wh, const __bf16* __restrict__ Wl,
            float* __restrict__ C, int M, int N, int K) {
  __shared__ __align__(16) __bf16 smem[2][4][128 * 32];   // [buf][Ah,Al,Wh,Wl]
  const int tid  = threadIdx.x;
  const int lane = tid & 31;
  const int wave = tid >> 5;
  const int ln = lane & 15, hf = lane >> 4;
  const int mblk = blockIdx.x * 128;
  const int nblk = blockIdx.y * 128;
  const int wm = wave >> 2, wn = wave & 3;
  (void)M;

  const unsigned sbase = (unsigned)(size_t)&smem[0][0][0];
  const int crow = tid >> 2;   // chunk row 0..63 (rep adds 64)
  const int c16  = tid & 3;    // 16B chunk within 64B row

  const __bf16* srcs[4] = {Ah + (long)mblk * K, Al + (long)mblk * K,
                           Wh + (long)nblk * K, Wl + (long)nblk * K};

  auto stage = [&](int buf, int kc) {
#pragma unroll
    for (int w = 0; w < 4; ++w) {
#pragma unroll
      for (int rep = 0; rep < 2; ++rep) {
        const int r = crow + rep * 64;
        const unsigned loff =
            sbase + (unsigned)(((buf * 4 + w) * 4096 + r * 32 + c16 * 8) * 2);
        const __bf16* g = srcs[w] + (long)r * K + kc + c16 * 8;
        asm volatile("global_load_async_to_lds_b128 %0, %1, off"
                     :: "v"(loff), "v"(g) : "memory");
      }
    }
  };

  v8f zero8 = {};
  v8f acc[4][2];
#pragma unroll
  for (int i = 0; i < 4; ++i)
#pragma unroll
    for (int j = 0; j < 2; ++j) acc[i][j] = zero8;

  const int nk = K / 32;
  stage(0, 0);
  for (int it = 0; it < nk; ++it) {
    if (it + 1 < nk) {
      stage((it + 1) & 1, (it + 1) * 32);
      asm volatile("s_wait_asynccnt 8" ::: "memory");   // prev stage retired
    } else {
      asm volatile("s_wait_asynccnt 0" ::: "memory");
    }
    __syncthreads();   // whole tile visible to all waves

    const int buf = it & 1;
    const __bf16* sA_h = &smem[buf][0][0];
    const __bf16* sA_l = &smem[buf][1][0];
    const __bf16* sB_h = &smem[buf][2][0];
    const __bf16* sB_l = &smem[buf][3][0];

    v16bf ah[4], al[4], bh[2], bl[2];
#pragma unroll
    for (int i = 0; i < 4; ++i) {
      const int r0 = (wm * 64 + 16 * i) * 32;
      ah[i] = ldA(sA_h + r0, 32, lane);
      al[i] = ldA(sA_l + r0, 32, lane);
    }
#pragma unroll
    for (int j = 0; j < 2; ++j) {
      const int r0 = (wn * 32 + 16 * j) * 32;
      bh[j] = ldB(sB_h + r0, 32, lane);
      bl[j] = ldB(sB_l + r0, 32, lane);
    }
#pragma unroll
    for (int i = 0; i < 4; ++i)
#pragma unroll
      for (int j = 0; j < 2; ++j) {
        acc[i][j] = wmma_bf16(ah[i], bh[j], acc[i][j]);
        acc[i][j] = wmma_bf16(ah[i], bl[j], acc[i][j]);
        acc[i][j] = wmma_bf16(al[i], bh[j], acc[i][j]);
      }
    __syncthreads();   // protect buf against next-next stage overwrite
  }

#pragma unroll
  for (int i = 0; i < 4; ++i)
#pragma unroll
    for (int j = 0; j < 2; ++j)
#pragma unroll
      for (int r = 0; r < 8; ++r)
        C[(long)(mblk + wm * 64 + 16 * i + r + 8 * hf) * N + nblk + wn * 32 +
          16 * j + ln] = acc[i][j][r];
}

// ------------------------------- flash attention ---------------------------
// One wave owns a 16-row q tile; k in 32-wide tiles. Q fragments are loaded
// once and held in registers for the whole k loop.

__global__ void __launch_bounds__(128)
flash_attn(const __bf16* __restrict__ qh, const __bf16* __restrict__ ql,
           const __bf16* __restrict__ kh, const __bf16* __restrict__ kl,
           const __bf16* __restrict__ vth, const __bf16* __restrict__ vtl,
           float* __restrict__ ao) {
  __shared__ __align__(16) __bf16 pstage[4][2][16 * 32];

  const int lane = threadIdx.x & 31;
  const int wave = threadIdx.x >> 5;
  const int ln = lane & 15, hf = lane >> 4;
  const int q0 = blockIdx.x * 64 + wave * 16;
  const int b  = (int)blockIdx.y >> 4;
  const int h  = (int)blockIdx.y & 15;
  const float scale = 0.08838834764831845f;   // 1/sqrt(128)

  __bf16* ps0 = &pstage[wave][0][0];
  __bf16* ps1 = &pstage[wave][1][0];

  const long qtile0 = (long)(b * Tc + q0) * Dc + h * HDc;
  const long vbase0 = (long)(b * Hc + h) * HDc * Tc;

  // Persistent Q fragments (16x128 hi/lo)
  v16bf aqh[4], aql[4];
#pragma unroll
  for (int d4 = 0; d4 < 4; ++d4) {
    aqh[d4] = ldA(qh + qtile0 + 32 * d4, Dc, lane);
    aql[d4] = ldA(ql + qtile0 + 32 * d4, Dc, lane);
  }

  v8f zero8 = {};
  v8f acc_o[8];
#pragma unroll
  for (int n = 0; n < 8; ++n) acc_o[n] = zero8;
  float m_r[8], l_r[8];
#pragma unroll
  for (int r = 0; r < 8; ++r) { m_r[r] = -3.0e38f; l_r[r] = 0.0f; }

  const int ktiles = (q0 + 16 + 31) >> 5;
  for (int kt = 0; kt < ktiles; ++kt) {
    const int kbase = kt * 32;

    // ---- scores S = Q K^T  (two 16x16 frags; bf16x3) ----
    v8f s[2] = {zero8, zero8};
#pragma unroll
    for (int d4 = 0; d4 < 4; ++d4) {
      const int dc = 32 * d4;
#pragma unroll
      for (int j = 0; j < 2; ++j) {
        const long koff = (long)(b * Tc + kbase + 16 * j) * Dc + h * HDc + dc;
        v16bf bkh = ldB(kh + koff, Dc, lane);
        v16bf bkl = ldB(kl + koff, Dc, lane);
        s[j] = wmma_bf16(aqh[d4], bkh, s[j]);
        s[j] = wmma_bf16(aqh[d4], bkl, s[j]);
        s[j] = wmma_bf16(aql[d4], bkh, s[j]);
      }
    }

    // ---- online softmax update ----
    float p0[8], p1[8];
#pragma unroll
    for (int r = 0; r < 8; ++r) {
      const int gi = q0 + r + 8 * hf;
      const int c0 = kbase + ln;
      const int c1 = kbase + 16 + ln;
      float s0 = s[0][r] * scale + (c0 <= gi ? 0.0f : -1.0e9f);
      float s1 = s[1][r] * scale + (c1 <= gi ? 0.0f : -1.0e9f);
      float mx = fmaxf(s0, s1);
#pragma unroll
      for (int off = 8; off; off >>= 1)
        mx = fmaxf(mx, __shfl_xor(mx, off, 32));
      float mnew  = fmaxf(m_r[r], mx);
      float alpha = __expf(m_r[r] - mnew);
      float e0 = __expf(s0 - mnew);
      float e1 = __expf(s1 - mnew);
      float sum = e0 + e1;
#pragma unroll
      for (int off = 8; off; off >>= 1)
        sum += __shfl_xor(sum, off, 32);
      l_r[r] = l_r[r] * alpha + sum;
      m_r[r] = mnew;
#pragma unroll
      for (int n = 0; n < 8; ++n) acc_o[n][r] *= alpha;
      p0[r] = e0;
      p1[r] = e1;
    }

    // ---- stage P (hi/lo bf16) to LDS, reload as A-fragment ----
#pragma unroll
    for (int r = 0; r < 8; ++r) {
      const int row = r + 8 * hf;
      __bf16 h0 = (__bf16)p0[r];
      ps0[row * 32 + ln] = h0;
      ps1[row * 32 + ln] = (__bf16)(p0[r] - (float)h0);
      __bf16 h1 = (__bf16)p1[r];
      ps0[row * 32 + 16 + ln] = h1;
      ps1[row * 32 + 16 + ln] = (__bf16)(p1[r] - (float)h1);
    }
    asm volatile("s_wait_dscnt 0" ::: "memory");   // same-wave DS in-order
    v16bf pf_h = ldA(ps0, 32, lane);
    v16bf pf_l = ldA(ps1, 32, lane);
    asm volatile("" ::: "memory");

    // ---- O += P * V ----
#pragma unroll
    for (int n = 0; n < 8; ++n) {
      const long voff = vbase0 + (long)(16 * n) * Tc + kbase;
      v16bf bvh = ldB(vth + voff, Tc, lane);
      v16bf bvl = ldB(vtl + voff, Tc, lane);
      acc_o[n] = wmma_bf16(pf_h, bvh, acc_o[n]);
      acc_o[n] = wmma_bf16(pf_h, bvl, acc_o[n]);
      acc_o[n] = wmma_bf16(pf_l, bvh, acc_o[n]);
    }
  }

#pragma unroll
  for (int r = 0; r < 8; ++r) {
    const float inv = 1.0f / l_r[r];
    const long rowoff = (long)(b * Tc + q0 + r + 8 * hf) * Dc + h * HDc;
#pragma unroll
    for (int n = 0; n < 8; ++n)
      ao[rowoff + 16 * n + ln] = acc_o[n][r] * inv;
  }
}

// ------------------------------- host launcher -----------------------------

extern "C" void kernel_launch(void* const* d_in, const int* in_sizes, int n_in,
                              void* d_out, int out_size, void* d_ws,
                              size_t ws_size, hipStream_t stream) {
  (void)in_sizes; (void)n_in; (void)out_size; (void)ws_size;

  const float* x    = (const float*)d_in[0];
  const float* cosv = (const float*)d_in[1];
  const float* sinv = (const float*)d_in[2];
  /* d_in[3] = mask: causal mask applied analytically (-1e9), unused */
  const float* wq   = (const float*)d_in[4];
  const float* wk   = (const float*)d_in[5];
  const float* wv   = (const float*)d_in[6];
  const float* wo   = (const float*)d_in[7];

  float* out_o = (float*)d_out;
  float* out_k = out_o + BTDc;
  float* out_v = out_o + 2 * BTDc;

  char* p = (char*)d_ws;
  auto take = [&](size_t bytes) {
    char* r = p;
    p += (bytes + 255) & ~(size_t)255;
    return r;
  };
  __bf16* x_hi = (__bf16*)take(BTDc * 2);
  __bf16* x_lo = (__bf16*)take(BTDc * 2);
  __bf16* wq_h = (__bf16*)take(DDc * 2);
  __bf16* wq_l = (__bf16*)take(DDc * 2);
  __bf16* wk_h = (__bf16*)take(DDc * 2);
  __bf16* wk_l = (__bf16*)take(DDc * 2);
  __bf16* wv_h = (__bf16*)take(DDc * 2);
  __bf16* wv_l = (__bf16*)take(DDc * 2);
  __bf16* wo_h = (__bf16*)take(DDc * 2);
  __bf16* wo_l = (__bf16*)take(DDc * 2);
  float*  qbuf = (float*)take(BTDc * 4);
  __bf16* q_h  = (__bf16*)take(BTDc * 2);
  __bf16* q_l  = (__bf16*)take(BTDc * 2);
  __bf16* k_h  = (__bf16*)take(BTDc * 2);
  __bf16* k_l  = (__bf16*)take(BTDc * 2);
  __bf16* vt_h = (__bf16*)take(BTDc * 2);
  __bf16* vt_l = (__bf16*)take(BTDc * 2);
  float*  ao   = qbuf;   // reuse: flash reads q_h/q_l, not qbuf
  __bf16* ao_h = x_hi;   // reuse: x splits dead after projections
  __bf16* ao_l = x_lo;

  const int thr = 256;

  split_kernel<<<(int)(BTDc / thr), thr, 0, stream>>>(x, x_hi, x_lo, BTDc);
  split_kernel<<<(int)(DDc / thr), thr, 0, stream>>>(wq, wq_h, wq_l, DDc);
  split_kernel<<<(int)(DDc / thr), thr, 0, stream>>>(wk, wk_h, wk_l, DDc);
  split_kernel<<<(int)(DDc / thr), thr, 0, stream>>>(wv, wv_h, wv_l, DDc);
  split_kernel<<<(int)(DDc / thr), thr, 0, stream>>>(wo, wo_h, wo_l, DDc);

  dim3 gg(Bc * Tc / 128, Dc / 128);   // (32,16)
  gemm_bf16x3<<<gg, 256, 0, stream>>>(x_hi, x_lo, wq_h, wq_l, qbuf,
                                      Bc * Tc, Dc, Dc);
  gemm_bf16x3<<<gg, 256, 0, stream>>>(x_hi, x_lo, wk_h, wk_l, out_k,
                                      Bc * Tc, Dc, Dc);
  gemm_bf16x3<<<gg, 256, 0, stream>>>(x_hi, x_lo, wv_h, wv_l, out_v,
                                      Bc * Tc, Dc, Dc);

  const long nrope = (long)Bc * Tc * Hc * (HDc / 2);
  rope_kernel<<<(int)(nrope / thr), thr, 0, stream>>>(qbuf, cosv, sinv);
  rope_kernel<<<(int)(nrope / thr), thr, 0, stream>>>(out_k, cosv, sinv);

  split_kernel<<<(int)(BTDc / thr), thr, 0, stream>>>(qbuf, q_h, q_l, BTDc);
  split_kernel<<<(int)(BTDc / thr), thr, 0, stream>>>(out_k, k_h, k_l, BTDc);
  splitT_kernel<<<(int)(BTDc / thr), thr, 0, stream>>>(out_v, vt_h, vt_l);

  flash_attn<<<dim3(Tc / 64, Bc * Hc), 128, 0, stream>>>(q_h, q_l, k_h, k_l,
                                                         vt_h, vt_l, ao);

  split_kernel<<<(int)(BTDc / thr), thr, 0, stream>>>(ao, ao_h, ao_l, BTDc);
  gemm_bf16x3<<<gg, 256, 0, stream>>>(ao_h, ao_l, wo_h, wo_l, out_o,
                                      Bc * Tc, Dc, Dc);
}
